// FullFastSTU_46858093199678
// MI455X (gfx1250) — matrix operand order repaired and checked by reading, not compiled, not run on hardware
//
#include <hip/hip_runtime.h>
#include <hip/hip_bf16.h>

typedef __attribute__((ext_vector_type(16))) __bf16 v16bf;
typedef __attribute__((ext_vector_type(8)))  __bf16 v8bf;
typedef __attribute__((ext_vector_type(2)))  __bf16 v2bf;
typedef __attribute__((ext_vector_type(8)))  float  v8f;

#define L_SEQ 2048
#define D_CH  512
#define SD_N  512
#define TK    48      // 2K
#define T_CH  64      // chunk length
#define NC    32      // L / T_CH
#define XW    96      // padded x-window rows (80 real + 16 zero)
#define KDIM  608     // 512 state rows + 96 x rows (19 * 32)
#define NJK   3072    // T_CH * TK
#define KD    24576   // TK * D_CH

// ---------------------------------------------------------------------------
// K1a: per-state precompute: A powers, coefT (chunk-summary weights),
//      powT = A^64, and the state rows of W^T:  Wt[(j*48+k)][s] = C[s,k]*A^{j+1}
// ---------------------------------------------------------------------------
__global__ void k_powpre(const float* __restrict__ A, const float* __restrict__ Bi,
                         const float* __restrict__ C, __bf16* __restrict__ Wt,
                         float* __restrict__ coefT, float* __restrict__ powT)
{
    int s = blockIdx.x * blockDim.x + threadIdx.x;
    if (s >= SD_N) return;
    float a = A[s], b = Bi[s];
    float p = 1.f;
    for (int j = 0; j < T_CH; ++j) {        // coefT[i][s] = B*A^{63-i}
        coefT[(T_CH - 1 - j) * SD_N + s] = b * p;
        p *= a;
    }
    powT[s] = p;                            // A^64
    float pj = a;
    for (int j = 0; j < T_CH; ++j) {
        for (int k = 0; k < TK; ++k)
            Wt[(size_t)(j * TK + k) * KDIM + s] = (__bf16)(C[s * TK + k] * pj);
        pj *= a;
    }
}

// ---------------------------------------------------------------------------
// K1b: truncated impulse response F[k][dt] = sum_s C[s,k]*B[s]*A[s]^dt
// ---------------------------------------------------------------------------
__global__ __launch_bounds__(256) void k_F(const float* __restrict__ A,
                                           const float* __restrict__ Bi,
                                           const float* __restrict__ C,
                                           float* __restrict__ F)
{
    __shared__ float red[256];
    int k  = blockIdx.x / T_CH;
    int dt = blockIdx.x % T_CH;
    int tid = threadIdx.x;
    float acc = 0.f;
    for (int s = tid; s < SD_N; s += 256)
        acc += C[s * TK + k] * Bi[s] * powf(A[s], (float)dt);
    red[tid] = acc;
    __syncthreads();
    for (int off = 128; off > 0; off >>= 1) {
        if (tid < off) red[tid] += red[tid + off];
        __syncthreads();
    }
    if (tid == 0) F[k * T_CH + dt] = red[0];
}

// ---------------------------------------------------------------------------
// K1c: x-window rows of W^T (intra-chunk LDS response + direct-conv M taps,
//      incl. 16-sample cross-chunk tail of M; rows >= 80 are zero padding)
// ---------------------------------------------------------------------------
__global__ void k_Wx(const float* __restrict__ F, const float* __restrict__ M,
                     __bf16* __restrict__ Wt)
{
    int idx = blockIdx.x * blockDim.x + threadIdx.x;
    if (idx >= NJK * XW) return;
    int col = idx / XW, i = idx % XW;
    int j = col / TK, k = col % TK;
    float val = 0.f;
    if (i < 80) {
        if (i >= 16) {                      // intra-chunk x sample
            int dt = j - (i - 16);
            if (dt >= 0) {
                val = F[k * T_CH + dt];
                if (dt < 16) val += M[k * 16 + dt];
            }
        } else {                            // previous-chunk tail (M only)
            int dtp = j + 16 - i;
            if (dtp <= 15) val = M[k * 16 + dtp];
        }
    }
    Wt[(size_t)col * KDIM + SD_N + i] = (__bf16)val;
}

// ---------------------------------------------------------------------------
// K1d: x -> transposed bf16  xT[d][t]
// ---------------------------------------------------------------------------
__global__ void k_xT(const float* __restrict__ x, __bf16* __restrict__ xT)
{
    int idx = blockIdx.x * blockDim.x + threadIdx.x;   // < 1048576
    int t = idx >> 9, d = idx & 511;
    xT[(size_t)d * L_SEQ + t] = (__bf16)x[idx];
}

// ---------------------------------------------------------------------------
// K1e: M_phi_{plus,minus} -> bf16, transposed+concatenated  Mc[o][k*512+d]
// ---------------------------------------------------------------------------
__global__ void k_mcat(const float* __restrict__ Mp, const float* __restrict__ Mm,
                       __bf16* __restrict__ Mc)
{
    int idx = blockIdx.x * blockDim.x + threadIdx.x;   // < 12582912
    int o = idx / KD;
    int r = idx % KD;
    int k = r >> 9, d = r & 511;
    float v = (k < 24) ? Mp[((size_t)k * 512 + d) * 512 + o]
                       : Mm[((size_t)(k - 24) * 512 + d) * 512 + o];
    Mc[idx] = (__bf16)v;
}

// ---------------------------------------------------------------------------
// K2a: inter-chunk scan.  Per (n,s): w_c = sum_i coefT[i][s]*x[n,c*64+i];
//      AA[c][n][s] = h (state entering chunk c);  h = A^64*h + w_c
//      AA rows are KDIM=608 wide: cols 0..511 = state, 512..607 = x-window.
// ---------------------------------------------------------------------------
__global__ __launch_bounds__(256) void k_scan(const __bf16* __restrict__ xT,
                                              const float* __restrict__ coefT,
                                              const float* __restrict__ powT,
                                              __bf16* __restrict__ AA)
{
    int n = blockIdx.x;
    int s = blockIdx.y * 256 + threadIdx.x;
    const __bf16* xrow = xT + (size_t)n * L_SEQ;
    float aT = powT[s];
    float h = 0.f;
    for (int c = 0; c < NC; ++c) {
        AA[((size_t)c * D_CH + n) * KDIM + s] = (__bf16)h;
        float w = 0.f;
        const __bf16* xc = xrow + c * T_CH;
        #pragma unroll 8
        for (int i = 0; i < T_CH; ++i)
            w = fmaf((float)xc[i], coefT[i * SD_N + s], w);
        h = fmaf(h, aT, w);
    }
}

// ---------------------------------------------------------------------------
// K2b: zero-padded x-window rows of AA:  AA[c][n][512+i] = x[n, c*64-16+i]
//      (all bounds checks hoisted out of the GEMM hot loop)
// ---------------------------------------------------------------------------
__global__ void k_xe(const __bf16* __restrict__ xT, __bf16* __restrict__ AA)
{
    int idx = blockIdx.x * blockDim.x + threadIdx.x;   // < NC*D_CH*XW
    int i = idx % XW;
    int r = idx / XW;          // c*512 + n
    int n = r & 511;
    int c = r >> 9;
    int t = c * T_CH - 16 + i;
    __bf16 val = (__bf16)0.0f;
    if (i < 80 && t >= 0) val = xT[(size_t)n * L_SEQ + t];
    AA[(size_t)r * KDIM + SD_N + i] = val;
}

// ---------------------------------------------------------------------------
// K3: per-chunk GEMM:  U_c[n,(j,k)] = AA_c @ W   (bf16 WMMA, uniform K-loop)
//     Block = 8 waves, block tile 64n x 256col, wave tile 32n x 64col.
//     Epilogue writes transposed into U2[l][k*512+n] with packed b32 stores.
// ---------------------------------------------------------------------------
__global__ __launch_bounds__(256) void k_chunk_gemm(const __bf16* __restrict__ AA,
                                                    const __bf16* __restrict__ Wt,
                                                    __bf16* __restrict__ U2)
{
    const int lane = threadIdx.x & 31;
    const int wave = threadIdx.x >> 5;
    const int c  = blockIdx.z;
    const int nW = blockIdx.x * 64  + (wave >> 2) * 32;
    const int cW = blockIdx.y * 256 + (wave & 3) * 64;
    const int mrow = lane & 15;
    const bool hiL = lane >= 16;

    // lane-resolved base pointers (A: rows, B: cols), uniform across K-loop
    const __bf16* ap0 = AA + ((size_t)c * D_CH + nW + mrow) * KDIM + (hiL ? 8 : 0);
    const __bf16* ap1 = ap0 + (size_t)16 * KDIM;
    const __bf16* bp  = Wt + (size_t)(cW + mrow) * KDIM + (hiL ? 8 : 0);

    v8f acc[2][4] = {};

    for (int kk0 = 0; kk0 < KDIM; kk0 += 32) {
        v16bf a[2];
        {
            v8bf lo = *(const v8bf*)(ap0 + kk0);
            v8bf hi = *(const v8bf*)(ap0 + kk0 + 16);
            a[0] = __builtin_shufflevector(lo, hi, 0,1,2,3,4,5,6,7,
                                           8,9,10,11,12,13,14,15);
            lo = *(const v8bf*)(ap1 + kk0);
            hi = *(const v8bf*)(ap1 + kk0 + 16);
            a[1] = __builtin_shufflevector(lo, hi, 0,1,2,3,4,5,6,7,
                                           8,9,10,11,12,13,14,15);
        }
        v16bf b[4];
        #pragma unroll
        for (int q = 0; q < 4; ++q) {
            const __bf16* cp = bp + (size_t)(q * 16) * KDIM + kk0;
            __builtin_prefetch(cp + 32, 0, 0);
            v8bf lo = *(const v8bf*)cp;
            v8bf hi = *(const v8bf*)(cp + 16);
            b[q] = __builtin_shufflevector(lo, hi, 0,1,2,3,4,5,6,7,
                                           8,9,10,11,12,13,14,15);
        }
        #pragma unroll
        for (int r = 0; r < 2; ++r)
            #pragma unroll
            for (int q = 0; q < 4; ++q)
                acc[r][q] = __builtin_amdgcn_wmma_f32_16x16x32_bf16(
                    false, a[r], false, b[q], (short)0, acc[r][q], false, false);
    }

    // epilogue: D tile (VGPR v -> M=v/v+8, lane -> N) -> U2[l][k*512+n], packed pairs
    #pragma unroll
    for (int r = 0; r < 2; ++r) {
        int nb = nW + r * 16 + (hiL ? 8 : 0);
        #pragma unroll
        for (int q = 0; q < 4; ++q) {
            int col = cW + q * 16 + mrow;
            int j = col / TK, k = col - j * TK;
            size_t base = (size_t)(c * T_CH + j) * KD + (size_t)k * D_CH + nb;
            #pragma unroll
            for (int p = 0; p < 4; ++p) {
                v2bf w;
                w[0] = (__bf16)acc[r][q][2 * p];
                w[1] = (__bf16)acc[r][q][2 * p + 1];
                *(v2bf*)(U2 + base + 2 * p) = w;
            }
        }
    }
}

// ---------------------------------------------------------------------------
// K4: spectral GEMM: out[l][o] = U2(2048 x 24576) @ Mc^T   (bf16 WMMA)
//     Block = 4 waves (2x2), wave tile 32l x 32o, K-loop 768 steps.
// ---------------------------------------------------------------------------
__global__ __launch_bounds__(128) void k_spectral(const __bf16* __restrict__ U2,
                                                  const __bf16* __restrict__ Mc,
                                                  __bf16* __restrict__ out)
{
    const int lane = threadIdx.x & 31;
    const int wave = threadIdx.x >> 5;          // 0..3
    const int lW = blockIdx.x * 64 + (wave >> 1) * 32;
    const int oW = blockIdx.y * 64 + (wave & 1) * 32;
    const int mrow = lane & 15;
    const bool hiL = lane >= 16;

    const __bf16* ap0 = U2 + (size_t)(lW + mrow) * KD + (hiL ? 8 : 0);
    const __bf16* ap1 = ap0 + (size_t)16 * KD;
    const __bf16* bp0 = Mc + (size_t)(oW + mrow) * KD + (hiL ? 8 : 0);
    const __bf16* bp1 = bp0 + (size_t)16 * KD;

    v8f acc[2][2] = {};

    for (int kk0 = 0; kk0 < KD; kk0 += 32) {
        v16bf a[2], b[2];
        {
            v8bf lo = *(const v8bf*)(ap0 + kk0);
            v8bf hi = *(const v8bf*)(ap0 + kk0 + 16);
            a[0] = __builtin_shufflevector(lo, hi, 0,1,2,3,4,5,6,7,
                                           8,9,10,11,12,13,14,15);
            lo = *(const v8bf*)(ap1 + kk0);
            hi = *(const v8bf*)(ap1 + kk0 + 16);
            a[1] = __builtin_shufflevector(lo, hi, 0,1,2,3,4,5,6,7,
                                           8,9,10,11,12,13,14,15);
            lo = *(const v8bf*)(bp0 + kk0);
            hi = *(const v8bf*)(bp0 + kk0 + 16);
            b[0] = __builtin_shufflevector(lo, hi, 0,1,2,3,4,5,6,7,
                                           8,9,10,11,12,13,14,15);
            lo = *(const v8bf*)(bp1 + kk0);
            hi = *(const v8bf*)(bp1 + kk0 + 16);
            b[1] = __builtin_shufflevector(lo, hi, 0,1,2,3,4,5,6,7,
                                           8,9,10,11,12,13,14,15);
        }
        #pragma unroll
        for (int r = 0; r < 2; ++r)
            #pragma unroll
            for (int q = 0; q < 2; ++q)
                acc[r][q] = __builtin_amdgcn_wmma_f32_16x16x32_bf16(
                    false, a[r], false, b[q], (short)0, acc[r][q], false, false);
    }

    #pragma unroll
    for (int r = 0; r < 2; ++r)
        #pragma unroll
        for (int q = 0; q < 2; ++q) {
            int o = oW + q * 16 + mrow;
            #pragma unroll
            for (int v = 0; v < 8; ++v) {
                int m = v + (hiL ? 8 : 0);
                int l = lW + r * 16 + m;
                out[(size_t)l * D_CH + o] = (__bf16)acc[r][q][v];
            }
        }
}

// ---------------------------------------------------------------------------
extern "C" void kernel_launch(void* const* d_in, const int* in_sizes, int n_in,
                              void* d_out, int out_size, void* d_ws, size_t ws_size,
                              hipStream_t stream)
{
    const float* x  = (const float*)d_in[0];
    // d_in[1] = input_pos (unused: positions are simply 0..L-1)
    const float* A  = (const float*)d_in[2];
    const float* Bi = (const float*)d_in[3];
    const float* C  = (const float*)d_in[4];
    const float* M  = (const float*)d_in[5];
    const float* Mp = (const float*)d_in[6];
    const float* Mm = (const float*)d_in[7];
    __bf16* out = (__bf16*)d_out;

    char* ws = (char*)d_ws;
    size_t off = 0;
    auto wsalloc = [&](size_t bytes) -> void* {
        void* p = (void*)(ws + off);
        off = (off + bytes + 255) & ~(size_t)255;
        return p;
    };
    __bf16* Wt    = (__bf16*)wsalloc((size_t)NJK * KDIM * 2);       // 3.6 MB
    float*  coefT = (float*) wsalloc((size_t)T_CH * SD_N * 4);      // 128 KB
    float*  powT  = (float*) wsalloc((size_t)SD_N * 4);             // 2 KB
    float*  F     = (float*) wsalloc((size_t)TK * T_CH * 4);        // 12 KB
    __bf16* xT    = (__bf16*)wsalloc((size_t)D_CH * L_SEQ * 2);     // 2 MB
    __bf16* AA    = (__bf16*)wsalloc((size_t)NC * D_CH * KDIM * 2); // 19.9 MB
    __bf16* U2    = (__bf16*)wsalloc((size_t)L_SEQ * KD * 2);       // 96 MB
    __bf16* Mc    = (__bf16*)wsalloc((size_t)D_CH * KD * 2);        // 24 MB

    k_powpre<<<2, 256, 0, stream>>>(A, Bi, C, Wt, coefT, powT);
    k_F<<<TK * T_CH, 256, 0, stream>>>(A, Bi, C, F);
    k_Wx<<<(NJK * XW) / 256, 256, 0, stream>>>(F, M, Wt);
    k_xT<<<(L_SEQ * D_CH) / 256, 256, 0, stream>>>(x, xT);
    k_mcat<<<(D_CH * KD) / 256, 256, 0, stream>>>(Mp, Mm, Mc);
    k_scan<<<dim3(D_CH, 2), 256, 0, stream>>>(xT, coefT, powT, AA);
    k_xe<<<(NC * D_CH * XW) / 256, 256, 0, stream>>>(xT, AA);
    k_chunk_gemm<<<dim3(8, 12, NC), 256, 0, stream>>>(AA, Wt, U2);
    k_spectral<<<dim3(L_SEQ / 64, D_CH / 64), 128, 0, stream>>>(U2, Mc, out);
}
